// GroupedQueryAttention_Qwen_3_51178830299592
// MI455X (gfx1250) — matrix operand order repaired and checked by reading, not compiled, not run on hardware
//
#include <hip/hip_runtime.h>
#include <hip/hip_fp16.h>
#include <stdint.h>
#include <math.h>

typedef _Float16 v16h __attribute__((ext_vector_type(16)));
typedef _Float16 v8h  __attribute__((ext_vector_type(8)));
typedef float    v8f  __attribute__((ext_vector_type(8)));
typedef int      v4i  __attribute__((ext_vector_type(4)));

#define SEQ_LEN   2048
#define DMODEL    2048
#define NHEADS    32
#define NKV       8
#define DHEAD     128
#define EPS_RMS   1e-6f

// ---- gfx1250 async-to-LDS path (guarded; falls back to VGPR copies) -------
#if defined(__gfx1250__) && __has_builtin(__builtin_amdgcn_global_load_async_to_lds_b128) && __has_builtin(__builtin_amdgcn_s_wait_asynccnt)
#define USE_ASYNC_LDS 1
typedef __attribute__((address_space(1))) v4i* as1_v4i;
typedef __attribute__((address_space(3))) v4i* as3_v4i;

__device__ __forceinline__ void async_copy_b128(const _Float16* gsrc, _Float16* lds_dst) {
  v4i* g = (v4i*)const_cast<_Float16*>(gsrc);   // generic typed ptr
  v4i* l = (v4i*)lds_dst;                       // generic typed ptr (to LDS object)
  __builtin_amdgcn_global_load_async_to_lds_b128((as1_v4i)g, (as3_v4i)l, 0, 0);
}
#else
#define USE_ASYNC_LDS 0
#endif

// ---------------------------------------------------------------------------
// Fused QKV projection GEMM (WMMA f16->f32) + optional RMSNorm + RoPE epilogue.
// Block: 256 threads (8 waves). Each block computes a 16-row x 128-col strip,
// i.e. one full head-dim slice, so the norm reduction is block-local.
// ---------------------------------------------------------------------------
template<bool DO_NORM_ROPE>
__global__ __launch_bounds__(256)
void qkv_proj_kernel(const float* __restrict__ x,
                     const float* __restrict__ W, int Ncols,
                     const float* __restrict__ scale,
                     const float* __restrict__ cosv,
                     const float* __restrict__ sinv,
                     float* __restrict__ out32,      // may be null (Q path)
                     _Float16* __restrict__ out16)   // [head][SEQ][128]
{
  __shared__ _Float16 lds_a[16][32];
  __shared__ _Float16 lds_b[32][128];
  __shared__ float    lds_c[16][128];
  __shared__ float    lds_rms[16];

  const int tid  = threadIdx.x;
  const int lane = tid & 31;
  const int wv   = tid >> 5;        // wave id 0..7 -> 16-col sub-tile
  const int head = blockIdx.y;
  const int M0   = blockIdx.x * 16;
  const int N0   = head * DHEAD;
  const int m    = lane & 15;
  const int kh   = lane >> 4;

  v8f acc = {};

  for (int kt = 0; kt < DMODEL / 32; ++kt) {
    // ---- stage A tile: x[M0..+15][kt*32..+31], f32 -> f16 in LDS
    {
      int idx = tid * 2;
      int r = idx >> 5, c = idx & 31;
      const float2 f = *(const float2*)&x[(size_t)(M0 + r) * DMODEL + kt * 32 + c];
      lds_a[r][c]     = (_Float16)f.x;
      lds_a[r][c + 1] = (_Float16)f.y;
    }
    // ---- stage B tile: W[kt*32..+31][N0..+127], f32 -> f16 in LDS (coalesced float4)
    #pragma unroll
    for (int i = 0; i < 4; ++i) {
      int base = (tid + 256 * i) * 4;
      int r = base >> 7, c = base & 127;
      const float4 f = *(const float4*)&W[(size_t)(kt * 32 + r) * Ncols + N0 + c];
      lds_b[r][c]     = (_Float16)f.x;
      lds_b[r][c + 1] = (_Float16)f.y;
      lds_b[r][c + 2] = (_Float16)f.z;
      lds_b[r][c + 3] = (_Float16)f.w;
    }
    if (kt + 1 < DMODEL / 32)
      __builtin_prefetch(&W[(size_t)((kt + 1) * 32) * Ncols + N0 + (tid & 127)], 0, 1);
    __syncthreads();

    // ---- A fragment: row = m; K runs [16*kh, +16)
    v16h a;
    {
      v8h lo = *(const v8h*)&lds_a[m][16 * kh];
      v8h hi = *(const v8h*)&lds_a[m][16 * kh + 8];
      #pragma unroll
      for (int e = 0; e < 8; ++e) { a[e] = lo[e]; a[e + 8] = hi[e]; }
    }
    // ---- B fragment: col = wv*16+m; element e -> K = 16*kh + e
    v16h b;
    {
      const int col = wv * 16 + m;
      #pragma unroll
      for (int e = 0; e < 16; ++e) b[e] = lds_b[16 * kh + e][col];
    }
    acc = __builtin_amdgcn_wmma_f32_16x16x32_f16(false, a, false, b, (short)0, acc,
                                                 false, false);
    __syncthreads();
  }

  // spill accumulator tile into LDS (D layout: M = r + 8*kh, N = wv*16+m)
  #pragma unroll
  for (int r = 0; r < 8; ++r)
    lds_c[r + 8 * kh][wv * 16 + m] = acc[r];
  __syncthreads();

  if (DO_NORM_ROPE) {
    if (tid < 16) {
      float ss = 0.f;
      #pragma unroll 8
      for (int c = 0; c < DHEAD; ++c) { float h = lds_c[tid][c]; ss += h * h; }
      lds_rms[tid] = rsqrtf(ss * (1.0f / DHEAD) + EPS_RMS);
    }
    __syncthreads();
    #pragma unroll
    for (int i = 0; i < 8; ++i) {
      int idx = tid + 256 * i;
      int r = idx >> 7, c = idx & 127;
      lds_c[r][c] *= lds_rms[r] * scale[c];
    }
    __syncthreads();
  }

  #pragma unroll
  for (int i = 0; i < 8; ++i) {
    int idx = tid + 256 * i;
    int r = idx >> 7, c = idx & 127;
    int rowg = M0 + r;
    float val = lds_c[r][c];
    if (DO_NORM_ROPE) {
      float other = lds_c[r][c ^ 64];
      float rot = (c < 64) ? -other : other;
      val = val * cosv[rowg * DHEAD + c] + rot * sinv[rowg * DHEAD + c];
    }
    size_t o = ((size_t)head * SEQ_LEN + rowg) * DHEAD + c;
    out16[o] = (_Float16)val;
    if (out32) out32[o] = val;
  }
}

// ---------------------------------------------------------------------------
// Flash attention, one wave per (head, 16-row query tile). Causal GQA.
// Scores + PV both via v_wmma_f32_16x16x32_f16; online softmax in D-layout.
// V tile staged via GLOBAL_LOAD_ASYNC_TO_LDS when available.
// ---------------------------------------------------------------------------
__global__ __launch_bounds__(32)
void attn_kernel(const _Float16* __restrict__ q16,   // [32][SEQ][128]
                 const _Float16* __restrict__ k16,   // [8][SEQ][128]
                 const _Float16* __restrict__ vv16,  // [8][SEQ][128]
                 _Float16* __restrict__ ctx16)       // [SEQ][4096]
{
  __shared__ _Float16 lds_p[16][32];
  __shared__ _Float16 lds_v[32][128];

  const int lane = threadIdx.x;
  const int head = blockIdx.y;
  const int kv   = head >> 2;            // GROUP = 4
  const int M0   = blockIdx.x * 16;
  const int m    = lane & 15;
  const int kh   = lane >> 4;
  const float sm_scale = 0.08838834764831845f;  // 1/sqrt(128)

  // Q fragments: 4 chunks of K=32 along d (loaded once, reused each key tile)
  v16h qa[4];
  #pragma unroll
  for (int dc = 0; dc < 4; ++dc) {
    const _Float16* qr = &q16[((size_t)head * SEQ_LEN + (M0 + m)) * DHEAD + dc * 32];
    v8h lo = *(const v8h*)&qr[16 * kh];
    v8h hi = *(const v8h*)&qr[16 * kh + 8];
    #pragma unroll
    for (int e = 0; e < 8; ++e) { qa[dc][e] = lo[e]; qa[dc][e + 8] = hi[e]; }
  }

  v8f ctx[8];
  #pragma unroll
  for (int n = 0; n < 8; ++n)
    #pragma unroll
    for (int r = 0; r < 8; ++r) ctx[n][r] = 0.f;
  float mrow[8], lrow[8];
  #pragma unroll
  for (int r = 0; r < 8; ++r) { mrow[r] = -INFINITY; lrow[r] = 0.f; }

  const int kt_max = (M0 + 15) / 32;   // causal: last key tile index (inclusive)

  for (int kt = 0; kt <= kt_max; ++kt) {
    // stage V tile (32 keys x 128 f16 = 8KB) into LDS
#if USE_ASYNC_LDS
    #pragma unroll
    for (int i = 0; i < 16; ++i) {
      int idx = lane + 32 * i;              // 512 x b128 chunks
      int r = idx >> 4, cc = (idx & 15) * 8;
      async_copy_b128(&vv16[((size_t)kv * SEQ_LEN + kt * 32 + r) * DHEAD + cc],
                      &lds_v[r][cc]);
    }
    __builtin_amdgcn_s_wait_asynccnt(0);
#else
    #pragma unroll
    for (int i = 0; i < 16; ++i) {
      int idx = lane + 32 * i;              // 512 x v8h chunks
      int r = idx >> 4, cc = (idx & 15) * 8;
      *(v8h*)&lds_v[r][cc] =
          *(const v8h*)&vv16[((size_t)kv * SEQ_LEN + kt * 32 + r) * DHEAD + cc];
    }
#endif
    __syncthreads();

    // scores for keys [kt*32,+16) and [kt*32+16,+32): K^T fragments straight
    // from global (16 contiguous f16 along d per lane = one 32B load)
    v8f s0, s1;
    #pragma unroll
    for (int r = 0; r < 8; ++r) { s0[r] = 0.f; s1[r] = 0.f; }
    #pragma unroll
    for (int dc = 0; dc < 4; ++dc) {
      v16h b0 = *(const v16h*)&k16[((size_t)kv * SEQ_LEN + (kt * 32 + m)) * DHEAD +
                                   dc * 32 + 16 * kh];
      v16h b1 = *(const v16h*)&k16[((size_t)kv * SEQ_LEN + (kt * 32 + 16 + m)) * DHEAD +
                                   dc * 32 + 16 * kh];
      s0 = __builtin_amdgcn_wmma_f32_16x16x32_f16(false, qa[dc], false, b0, (short)0,
                                                  s0, false, false);
      s1 = __builtin_amdgcn_wmma_f32_16x16x32_f16(false, qa[dc], false, b1, (short)0,
                                                  s1, false, false);
    }

    // causal mask + scale + online softmax (row = M0 + r + 8*kh, key = lane&15)
    #pragma unroll
    for (int r = 0; r < 8; ++r) {
      int row  = M0 + r + 8 * kh;
      int key0 = kt * 32 + m;
      int key1 = key0 + 16;
      float a0 = (key0 <= row) ? s0[r] * sm_scale : -INFINITY;
      float a1 = (key1 <= row) ? s1[r] * sm_scale : -INFINITY;
      float mr = fmaxf(a0, a1);
      #pragma unroll
      for (int off = 1; off < 16; off <<= 1)
        mr = fmaxf(mr, __shfl_xor(mr, off, 32));
      float mnew = fmaxf(mrow[r], mr);
      float p0 = __expf(a0 - mnew);
      float p1 = __expf(a1 - mnew);
      float ps = p0 + p1;
      #pragma unroll
      for (int off = 1; off < 16; off <<= 1)
        ps += __shfl_xor(ps, off, 32);
      float corr = __expf(mrow[r] - mnew);
      lrow[r] = lrow[r] * corr + ps;
      mrow[r] = mnew;
      #pragma unroll
      for (int n = 0; n < 8; ++n) ctx[n][r] *= corr;
      lds_p[r + 8 * kh][m]      = (_Float16)p0;   // D-layout -> A-layout via LDS
      lds_p[r + 8 * kh][16 + m] = (_Float16)p1;
    }
    __syncthreads();

    // ctx += P(16x32) @ V(32x128)
    v16h pa;
    {
      v8h lo = *(const v8h*)&lds_p[m][16 * kh];
      v8h hi = *(const v8h*)&lds_p[m][16 * kh + 8];
      #pragma unroll
      for (int e = 0; e < 8; ++e) { pa[e] = lo[e]; pa[e + 8] = hi[e]; }
    }
    #pragma unroll
    for (int n = 0; n < 8; ++n) {
      v16h bv;
      #pragma unroll
      for (int e = 0; e < 16; ++e) bv[e] = lds_v[16 * kh + e][n * 16 + m];
      ctx[n] = __builtin_amdgcn_wmma_f32_16x16x32_f16(false, pa, false, bv, (short)0,
                                                      ctx[n], false, false);
    }
    __syncthreads();
  }

  #pragma unroll
  for (int n = 0; n < 8; ++n)
    #pragma unroll
    for (int r = 0; r < 8; ++r) {
      float val = ctx[n][r] / lrow[r];
      ctx16[(size_t)(M0 + r + 8 * kh) * (NHEADS * DHEAD) + head * DHEAD + n * 16 + m] =
          (_Float16)val;
    }
}

// ---------------------------------------------------------------------------
// Output projection: out = ctx16 (f16) @ Wo (f32->f16), f32 result.
// A tile (already f16 in ws) staged via async-to-LDS when available.
// ---------------------------------------------------------------------------
__global__ __launch_bounds__(256)
void out_proj_kernel(const _Float16* __restrict__ ctx16,  // [SEQ][4096]
                     const float* __restrict__ Wo,        // [4096][2048]
                     float* __restrict__ out)             // [SEQ][2048]
{
  __shared__ _Float16 lds_a[16][32];
  __shared__ _Float16 lds_b[32][128];

  const int tid  = threadIdx.x;
  const int lane = tid & 31;
  const int wv   = tid >> 5;
  const int M0   = blockIdx.x * 16;
  const int N0   = blockIdx.y * 128;
  const int m    = lane & 15;
  const int kh   = lane >> 4;
  const int KDIM = NHEADS * DHEAD;  // 4096
  const int NOUT = DMODEL;          // 2048

  v8f acc = {};
  for (int kt = 0; kt < KDIM / 32; ++kt) {
    // ---- A tile: 16x32 f16 = 64 b128 chunks
#if USE_ASYNC_LDS
    if (tid < 64) {
      int r = tid >> 2, cc = (tid & 3) * 8;
      async_copy_b128(&ctx16[(size_t)(M0 + r) * KDIM + kt * 32 + cc], &lds_a[r][cc]);
    }
    __builtin_amdgcn_s_wait_asynccnt(0);
#else
    {
      int r = tid >> 4, c = (tid & 15) * 2;
      *(uint32_t*)&lds_a[r][c] =
          *(const uint32_t*)&ctx16[(size_t)(M0 + r) * KDIM + kt * 32 + c];
    }
#endif
    #pragma unroll
    for (int i = 0; i < 4; ++i) {
      int base = (tid + 256 * i) * 4;
      int r = base >> 7, c = base & 127;
      const float4 f = *(const float4*)&Wo[(size_t)(kt * 32 + r) * NOUT + N0 + c];
      lds_b[r][c]     = (_Float16)f.x;
      lds_b[r][c + 1] = (_Float16)f.y;
      lds_b[r][c + 2] = (_Float16)f.z;
      lds_b[r][c + 3] = (_Float16)f.w;
    }
    __syncthreads();

    v16h a;
    {
      v8h lo = *(const v8h*)&lds_a[m][16 * kh];
      v8h hi = *(const v8h*)&lds_a[m][16 * kh + 8];
      #pragma unroll
      for (int e = 0; e < 8; ++e) { a[e] = lo[e]; a[e + 8] = hi[e]; }
    }
    v16h b;
    {
      const int col = wv * 16 + m;
      #pragma unroll
      for (int e = 0; e < 16; ++e) b[e] = lds_b[16 * kh + e][col];
    }
    acc = __builtin_amdgcn_wmma_f32_16x16x32_f16(false, a, false, b, (short)0, acc,
                                                 false, false);
    __syncthreads();
  }
  #pragma unroll
  for (int r = 0; r < 8; ++r)
    out[(size_t)(M0 + r + 8 * kh) * NOUT + N0 + wv * 16 + m] = acc[r];
}

// ---------------------------------------------------------------------------
extern "C" void kernel_launch(void* const* d_in, const int* in_sizes, int n_in,
                              void* d_out, int out_size, void* d_ws, size_t ws_size,
                              hipStream_t stream) {
  (void)in_sizes; (void)n_in; (void)out_size; (void)ws_size;
  const float* x  = (const float*)d_in[0];
  const float* Wq = (const float*)d_in[1];
  const float* Wk = (const float*)d_in[2];
  const float* Wv = (const float*)d_in[3];
  const float* Wo = (const float*)d_in[4];
  const float* qs = (const float*)d_in[5];
  const float* ks = (const float*)d_in[6];
  const float* cs = (const float*)d_in[7];
  const float* sn = (const float*)d_in[8];
  // d_in[9] = mask (causality computed analytically), d_in[10] = start_pos (0)

  float* out  = (float*)d_out;
  float* kout = out + (size_t)SEQ_LEN * DMODEL;
  float* vout = kout + (size_t)NKV * SEQ_LEN * DHEAD;

  _Float16* q16 = (_Float16*)d_ws;
  _Float16* k16 = q16 + (size_t)NHEADS * SEQ_LEN * DHEAD;
  _Float16* v16 = k16 + (size_t)NKV * SEQ_LEN * DHEAD;
  _Float16* c16 = v16 + (size_t)NKV * SEQ_LEN * DHEAD;

  dim3 blk(256);
  qkv_proj_kernel<true ><<<dim3(SEQ_LEN / 16, NHEADS), blk, 0, stream>>>(
      x, Wq, NHEADS * DHEAD, qs, cs, sn, nullptr, q16);
  qkv_proj_kernel<true ><<<dim3(SEQ_LEN / 16, NKV), blk, 0, stream>>>(
      x, Wk, NKV * DHEAD, ks, cs, sn, kout, k16);
  qkv_proj_kernel<false><<<dim3(SEQ_LEN / 16, NKV), blk, 0, stream>>>(
      x, Wv, NKV * DHEAD, ks, cs, sn, vout, v16);

  attn_kernel<<<dim3(SEQ_LEN / 16, NHEADS), dim3(32), 0, stream>>>(q16, k16, v16, c16);

  out_proj_kernel<<<dim3(SEQ_LEN / 16, DMODEL / 128), blk, 0, stream>>>(c16, Wo, out);
}
